// StitchingNet_9457517986041
// MI455X (gfx1250) — compile-verified, matrix operand order checked by reading, not run on hardware
//
#include <hip/hip_runtime.h>
#include <hip/hip_bf16.h>

typedef _Float16 v16h __attribute__((ext_vector_type(16)));
typedef _Float16 v8h  __attribute__((ext_vector_type(8)));
typedef float    v8f  __attribute__((ext_vector_type(8)));

#define BNT 256

// ---------------- direct conv (KxK, stride 1) + bias ----------------
__global__ void conv2d_k(const float* __restrict__ x, const float* __restrict__ w,
                         const float* __restrict__ bias, float* __restrict__ y,
                         int B, int Cin, int H, int W, int Cout, int K, int pad) {
    int Ho = H + 2 * pad - K + 1, Wo = W + 2 * pad - K + 1;
    long total = (long)B * Cout * Ho * Wo;
    long i = (long)blockIdx.x * blockDim.x + threadIdx.x;
    if (i >= total) return;
    int xw = (int)(i % Wo); long t = i / Wo;
    int yh = (int)(t % Ho); t /= Ho;
    int co = (int)(t % Cout); int b = (int)(t / Cout);
    float acc = bias[co];
    for (int ci = 0; ci < Cin; ++ci) {
        const float* xp = x + ((long)(b * Cin + ci)) * H * W;
        const float* wp = w + ((long)(co * Cin + ci)) * K * K;
        for (int ky = 0; ky < K; ++ky) {
            int iy = yh + ky - pad;
            if (iy < 0 || iy >= H) continue;
            for (int kx = 0; kx < K; ++kx) {
                int ix = xw + kx - pad;
                if (ix < 0 || ix >= W) continue;
                acc += xp[(long)iy * W + ix] * wp[ky * K + kx];
            }
        }
    }
    y[i] = acc;
}

// ---------------- batchnorm statistics: one block per channel ----------------
__global__ void bn_stats_k(const float* __restrict__ x, float* __restrict__ mean,
                           float* __restrict__ istd, int B, int C, int HW) {
    int c = blockIdx.x;
    int N = B * HW;
    float s = 0.f, s2 = 0.f;
    for (int i = threadIdx.x; i < N; i += BNT) {
        int b = i / HW, p = i % HW;
        float v = x[((long)(b * C + c)) * HW + p];
        s += v; s2 += v * v;
    }
    __shared__ float sh[BNT], sh2[BNT];
    sh[threadIdx.x] = s; sh2[threadIdx.x] = s2;
    __syncthreads();
    for (int off = BNT / 2; off > 0; off >>= 1) {
        if ((int)threadIdx.x < off) {
            sh[threadIdx.x]  += sh[threadIdx.x + off];
            sh2[threadIdx.x] += sh2[threadIdx.x + off];
        }
        __syncthreads();
    }
    if (threadIdx.x == 0) {
        float m = sh[0] / N;
        float v = sh2[0] / N - m * m;
        mean[c] = m;
        istd[c] = rsqrtf(v + 1e-5f);
    }
}

// ---------------- BN apply + ReLU (in place) ----------------
__global__ void bn_relu_k(float* __restrict__ x, const float* __restrict__ mean,
                          const float* __restrict__ istd, const float* __restrict__ g,
                          const float* __restrict__ bt, int B, int C, int HW) {
    long total = (long)B * C * HW;
    long i = (long)blockIdx.x * blockDim.x + threadIdx.x;
    if (i >= total) return;
    int c = (int)((i / HW) % C);
    float v = (x[i] - mean[c]) * istd[c] * g[c] + bt[c];
    x[i] = v > 0.f ? v : 0.f;
}

// ---------------- 2x2 maxpool stride 2 ----------------
__global__ void maxpool_k(const float* __restrict__ x, float* __restrict__ y,
                          int BC, int H, int W) {
    int Ho = H / 2, Wo = W / 2;
    long total = (long)BC * Ho * Wo;
    long i = (long)blockIdx.x * blockDim.x + threadIdx.x;
    if (i >= total) return;
    int ox = (int)(i % Wo); long t = i / Wo;
    int oy = (int)(t % Ho); int bc = (int)(t / Ho);
    const float* xp = x + (long)bc * H * W;
    int iy = 2 * oy, ix = 2 * ox;
    float m = fmaxf(fmaxf(xp[(long)iy * W + ix], xp[(long)iy * W + ix + 1]),
                    fmaxf(xp[(long)(iy + 1) * W + ix], xp[(long)(iy + 1) * W + ix + 1]));
    y[i] = m;
}

// ---------------- f32 [b][c][p] -> f16 [b][p_pad][c] transpose (zero-padded rows) ----------------
__global__ void f16t_k(const float* __restrict__ f, _Float16* __restrict__ ft,
                       int B, int C, int P, int Ppad) {
    long total = (long)B * Ppad * C;
    long i = (long)blockIdx.x * blockDim.x + threadIdx.x;
    if (i >= total) return;
    int c = (int)(i % C); long t = i / C;
    int p = (int)(t % Ppad); int b = (int)(t / Ppad);
    float v = (p < P) ? f[((long)(b * C + c)) * P + p] : 0.0f;
    ft[i] = (_Float16)v;
}

// ---- correlation GEMM: 4x2 register-blocked, 64x32 output tile per wave, f16 WMMA ----
__global__ void __launch_bounds__(128, 1)
corr_wmma_k(const _Float16* __restrict__ aT, const _Float16* __restrict__ bT,
            float* __restrict__ out, int C, int P, int Ppad) {
    int tilesQ = Ppad >> 5;                        // 32-wide q supertiles
    int tp = blockIdx.x;                           // 64-wide p supertile
    int tq = blockIdx.y * 4 + threadIdx.y;         // 4 waves per block, one q-supertile each
    if (tq >= tilesQ) return;                      // uniform per wave
    int b = blockIdx.z;
    int lane = threadIdx.x;                        // 0..31, wave32
    int r  = lane & 15;                            // M row (A) / N col (B)
    int kb = (lane < 16) ? 0 : 8;                  // K sub-offset per half-wave
    const long rowstride = (long)16 * C;
    const _Float16* abase = aT + ((long)b * Ppad + (long)tp * 64 + r) * C + kb;
    const _Float16* bbase = bT + ((long)b * Ppad + (long)tq * 32 + r) * C + kb;

    v8f vzero = {};
    v8f acc[4][2];
#pragma unroll
    for (int i = 0; i < 4; ++i)
#pragma unroll
        for (int j = 0; j < 2; ++j) acc[i][j] = vzero;

    for (int k = 0; k < C; k += 32) {
        union { v16h v; v8h h[2]; } A[4], Bm[2];
#pragma unroll
        for (int i = 0; i < 4; ++i) {
            A[i].h[0] = *(const v8h*)(abase + (long)i * rowstride + k);
            A[i].h[1] = *(const v8h*)(abase + (long)i * rowstride + k + 16);
        }
#pragma unroll
        for (int j = 0; j < 2; ++j) {
            Bm[j].h[0] = *(const v8h*)(bbase + (long)j * rowstride + k);
            Bm[j].h[1] = *(const v8h*)(bbase + (long)j * rowstride + k + 16);
        }
#pragma unroll
        for (int i = 0; i < 4; ++i)
#pragma unroll
            for (int j = 0; j < 2; ++j)
                acc[i][j] = __builtin_amdgcn_wmma_f32_16x16x32_f16(
                    false, A[i].v, false, Bm[j].v, (short)0, acc[i][j], false, false);
    }

    float* o = out + (long)b * P * P;
    int n = lane & 15;
    int mbase = (lane < 16) ? 0 : 8;
#pragma unroll
    for (int i = 0; i < 4; ++i) {
#pragma unroll
        for (int j = 0; j < 2; ++j) {
            int q = tq * 32 + j * 16 + n;
            if (q < P) {
#pragma unroll
                for (int rr = 0; rr < 8; ++rr) {
                    int p = tp * 64 + i * 16 + mbase + rr;
                    if (p < P) o[(long)p * P + q] = acc[i][j][rr];
                }
            }
        }
    }
}

// ---------------- column softmax (axis p), online max+sum, 2 passes, in place ----------------
__global__ void softmax_col_k(float* __restrict__ corr, int P) {
    int q = blockIdx.x * 32 + threadIdx.x;
    int b = blockIdx.y;
    float* m = corr + (long)b * P * P;
    __shared__ float redM[8][32], redS[8][32];
    float mx = -3.4e38f, s = 0.f;
    if (q < P) {
        for (int p = threadIdx.y; p < P; p += 8) {
            float v = m[(long)p * P + q];
            float nm = fmaxf(mx, v);
            s = s * __expf(mx - nm) + __expf(v - nm);
            mx = nm;
        }
    }
    redM[threadIdx.y][threadIdx.x] = mx;
    redS[threadIdx.y][threadIdx.x] = s;
    __syncthreads();
    if (threadIdx.y == 0) {
        float M = redM[0][threadIdx.x], S = redS[0][threadIdx.x];
        for (int j = 1; j < 8; ++j) {
            float mj = redM[j][threadIdx.x], sj = redS[j][threadIdx.x];
            float nm = fmaxf(M, mj);
            S = S * __expf(M - nm) + sj * __expf(mj - nm);
            M = nm;
        }
        redM[0][threadIdx.x] = M;
        redS[0][threadIdx.x] = 1.0f / S;
    }
    __syncthreads();
    mx = redM[0][threadIdx.x];
    float invs = redS[0][threadIdx.x];
    if (q < P)
        for (int p = threadIdx.y; p < P; p += 8)
            m[(long)p * P + q] = __expf(m[(long)p * P + q] - mx) * invs;
}

// ---------------- channel concat ----------------
__global__ void concat_k(const float* __restrict__ a, const float* __restrict__ b,
                         float* __restrict__ y, int B, int C, int HW) {
    long total = (long)B * 2 * C * HW;
    long i = (long)blockIdx.x * blockDim.x + threadIdx.x;
    if (i >= total) return;
    int p = (int)(i % HW); long t = i / HW;
    int c = (int)(t % (2 * C)); int bb = (int)(t / (2 * C));
    y[i] = (c < C) ? a[((long)(bb * C + c)) * HW + p]
                   : b[((long)(bb * C + (c - C))) * HW + p];
}

// ---------------- 1x1 conv (Cin->1) + sigmoid ----------------
__global__ void blend_k(const float* __restrict__ x, const float* __restrict__ w,
                        const float* __restrict__ bias, float* __restrict__ y,
                        int B, int C, int HW) {
    int total = B * HW;
    int i = blockIdx.x * blockDim.x + threadIdx.x;
    if (i >= total) return;
    int p = i % HW, b = i / HW;
    float acc = bias[0];
    for (int c = 0; c < C; ++c) acc += x[((long)(b * C + c)) * HW + p] * w[c];
    y[i] = 1.0f / (1.0f + __expf(-acc));
}

// ---------------- bilinear resize (half-pixel rule per reference) ----------------
__global__ void resize_k(const float* __restrict__ ws, float* __restrict__ wout,
                         int B, int IH, int IW, int OH, int OW) {
    int total = B * OH * OW;
    int i = blockIdx.x * blockDim.x + threadIdx.x;
    if (i >= total) return;
    int x = i % OW; int t = i / OW;
    int y = t % OH; int b = t / OH;
    float sy = fmaxf((y + 0.5f) * ((float)IH / OH) - 0.5f, 0.0f);
    int y0 = (int)floorf(sy); float fy = sy - (float)y0;
    int y1 = min(y0 + 1, IH - 1);
    float sx = fmaxf((x + 0.5f) * ((float)IW / OW) - 0.5f, 0.0f);
    int x0 = (int)floorf(sx); float fx = sx - (float)x0;
    int x1 = min(x0 + 1, IW - 1);
    const float* s = ws + (long)b * IH * IW;
    float top = s[y0 * IW + x0] * (1.f - fx) + s[y0 * IW + x1] * fx;
    float bot = s[y1 * IW + x0] * (1.f - fx) + s[y1 * IW + x1] * fx;
    wout[i] = top * (1.f - fy) + bot * fy;
}

// ---------------- final blend: result = w*img1 + (1-w)*img2 ----------------
__global__ void final_k(const float* __restrict__ img1, const float* __restrict__ img2,
                        const float* __restrict__ w, float* __restrict__ res,
                        int B, int Cimg, int HW) {
    int total = B * Cimg * HW;
    int i = blockIdx.x * blockDim.x + threadIdx.x;
    if (i >= total) return;
    int p = i % HW; int t = i / HW;
    int b = t / Cimg;
    float ww = w[(long)b * HW + p];
    res[i] = ww * img1[i] + (1.f - ww) * img2[i];
}

extern "C" void kernel_launch(void* const* d_in, const int* in_sizes, int n_in,
                              void* d_out, int out_size, void* d_ws, size_t ws_size,
                              hipStream_t stream) {
    (void)in_sizes; (void)n_in; (void)out_size; (void)ws_size;

    constexpr int B = 2;
    constexpr int H0 = 80;
    constexpr int C1 = 64,  H1 = 79, P1 = H1 * H1;      // 6241
    constexpr int Hq1 = 39, Pq1 = Hq1 * Hq1;            // 1521
    constexpr int C2 = 128, H2 = 38, P2 = H2 * H2;      // 1444
    constexpr int Hq2 = 19, Pq2 = Hq2 * Hq2;            // 361
    constexpr int C3 = 256, H3 = 18, P3 = H3 * H3;      // 324
    constexpr int PAD1 = 6272, PAD2 = 1472, PAD3 = 384; // P rounded up to 64

    // ----- inputs -----
    const float* img1 = (const float*)d_in[0];
    const float* img2 = (const float*)d_in[1];
    int pi = 2;
    const float* c1_w   = (const float*)d_in[pi++]; const float* c1_b   = (const float*)d_in[pi++];
    const float* bn1_g  = (const float*)d_in[pi++]; const float* bn1_b  = (const float*)d_in[pi++];
    const float* c2_w   = (const float*)d_in[pi++]; const float* c2_b   = (const float*)d_in[pi++];
    const float* bn2_g  = (const float*)d_in[pi++]; const float* bn2_b  = (const float*)d_in[pi++];
    const float* c3_w   = (const float*)d_in[pi++]; const float* c3_b   = (const float*)d_in[pi++];
    const float* bn3_g  = (const float*)d_in[pi++]; const float* bn3_b  = (const float*)d_in[pi++];
    const float* fw1    = (const float*)d_in[pi++]; const float* fb1    = (const float*)d_in[pi++];
    const float* fbn1_g = (const float*)d_in[pi++]; const float* fbn1_b = (const float*)d_in[pi++];
    const float* fw2    = (const float*)d_in[pi++]; const float* fb2    = (const float*)d_in[pi++];
    const float* fbn2_g = (const float*)d_in[pi++]; const float* fbn2_b = (const float*)d_in[pi++];
    const float* fw3    = (const float*)d_in[pi++]; const float* fb3    = (const float*)d_in[pi++];
    const float* fbn3_g = (const float*)d_in[pi++]; const float* fbn3_b = (const float*)d_in[pi++];
    const float* bl_w   = (const float*)d_in[pi++]; const float* bl_b   = (const float*)d_in[pi++];

    // ----- output regions (result, w, corr1, corr2, corr3) -----
    float* outf = (float*)d_out;
    const long O_W  = (long)B * 3 * H0 * H0;            // 38400
    const long O_C1 = O_W + (long)B * H0 * H0;          // 51200
    const long O_C2 = O_C1 + (long)B * P1 * P1;
    const long O_C3 = O_C2 + (long)B * P2 * P2;
    float* res_o  = outf;
    float* w_o    = outf + O_W;
    float* corr1o = outf + O_C1;
    float* corr2o = outf + O_C2;
    float* corr3o = outf + O_C3;

    // ----- workspace carve-out -----
    float* wsf = (float*)d_ws;
    size_t off = 0;
    auto alloc = [&](size_t n) { float* p = wsf + off; off += n; return p; };
    float* f1a = alloc((size_t)B * C1 * P1);  float* f1b = alloc((size_t)B * C1 * P1);
    float* p1a = alloc((size_t)B * C1 * Pq1); float* p1b = alloc((size_t)B * C1 * Pq1);
    float* f2a = alloc((size_t)B * C2 * P2);  float* f2b = alloc((size_t)B * C2 * P2);
    float* p2a = alloc((size_t)B * C2 * Pq2); float* p2b = alloc((size_t)B * C2 * Pq2);
    float* f3a = alloc((size_t)B * C3 * P3);  float* f3b = alloc((size_t)B * C3 * P3);
    float* fu0 = alloc((size_t)B * 512 * P3);
    float* fu1 = alloc((size_t)B * 256 * P3);
    float* fu2 = alloc((size_t)B * 128 * P3);
    float* fu3 = alloc((size_t)B * 64 * P3);
    float* blm = alloc((size_t)B * P3);
    float* mean = alloc(512); float* istd = alloc(512);
    _Float16* t1a = (_Float16*)alloc((size_t)B * PAD1 * C1 / 2);
    _Float16* t1b = (_Float16*)alloc((size_t)B * PAD1 * C1 / 2);
    _Float16* t2a = (_Float16*)alloc((size_t)B * PAD2 * C2 / 2);
    _Float16* t2b = (_Float16*)alloc((size_t)B * PAD2 * C2 / 2);
    _Float16* t3a = (_Float16*)alloc((size_t)B * PAD3 * C3 / 2);
    _Float16* t3b = (_Float16*)alloc((size_t)B * PAD3 * C3 / 2);

    auto NB = [](long n) { return (unsigned)((n + 255) / 256); };

    // ----- feature extractor (shared weights, run per image) -----
    auto feat = [&](const float* img, float* f1, float* p1, float* f2, float* p2, float* f3) {
        conv2d_k<<<NB((long)B * C1 * P1), 256, 0, stream>>>(img, c1_w, c1_b, f1, B, 3, H0, H0, C1, 4, 1);
        bn_stats_k<<<C1, BNT, 0, stream>>>(f1, mean, istd, B, C1, P1);
        bn_relu_k<<<NB((long)B * C1 * P1), 256, 0, stream>>>(f1, mean, istd, bn1_g, bn1_b, B, C1, P1);
        maxpool_k<<<NB((long)B * C1 * Pq1), 256, 0, stream>>>(f1, p1, B * C1, H1, H1);

        conv2d_k<<<NB((long)B * C2 * P2), 256, 0, stream>>>(p1, c2_w, c2_b, f2, B, C1, Hq1, Hq1, C2, 4, 1);
        bn_stats_k<<<C2, BNT, 0, stream>>>(f2, mean, istd, B, C2, P2);
        bn_relu_k<<<NB((long)B * C2 * P2), 256, 0, stream>>>(f2, mean, istd, bn2_g, bn2_b, B, C2, P2);
        maxpool_k<<<NB((long)B * C2 * Pq2), 256, 0, stream>>>(f2, p2, B * C2, H2, H2);

        conv2d_k<<<NB((long)B * C3 * P3), 256, 0, stream>>>(p2, c3_w, c3_b, f3, B, C2, Hq2, Hq2, C3, 4, 1);
        bn_stats_k<<<C3, BNT, 0, stream>>>(f3, mean, istd, B, C3, P3);
        bn_relu_k<<<NB((long)B * C3 * P3), 256, 0, stream>>>(f3, mean, istd, bn3_g, bn3_b, B, C3, P3);
    };
    feat(img1, f1a, p1a, f2a, p2a, f3a);
    feat(img2, f1b, p1b, f2b, p2b, f3b);

    // ----- correlation + softmax per level (WMMA path) -----
    auto corr = [&](const float* fa, const float* fb, _Float16* ta, _Float16* tb,
                    float* outr, int C, int P, int Ppad) {
        long tt = (long)B * Ppad * C;
        f16t_k<<<NB(tt), 256, 0, stream>>>(fa, ta, B, C, P, Ppad);
        f16t_k<<<NB(tt), 256, 0, stream>>>(fb, tb, B, C, P, Ppad);
        int tilesP = Ppad / 64, tilesQ = Ppad / 32;
        dim3 grid(tilesP, (tilesQ + 3) / 4, B);
        corr_wmma_k<<<grid, dim3(32, 4), 0, stream>>>(ta, tb, outr, C, P, Ppad);
        dim3 sg((P + 31) / 32, B);
        softmax_col_k<<<sg, dim3(32, 8), 0, stream>>>(outr, P);
    };
    corr(f1a, f1b, t1a, t1b, corr1o, C1, P1, PAD1);
    corr(f2a, f2b, t2a, t2b, corr2o, C2, P2, PAD2);
    corr(f3a, f3b, t3a, t3b, corr3o, C3, P3, PAD3);

    // ----- fusion head -----
    concat_k<<<NB((long)B * 512 * P3), 256, 0, stream>>>(f3a, f3b, fu0, B, C3, P3);

    conv2d_k<<<NB((long)B * 256 * P3), 256, 0, stream>>>(fu0, fw1, fb1, fu1, B, 512, H3, H3, 256, 3, 1);
    bn_stats_k<<<256, BNT, 0, stream>>>(fu1, mean, istd, B, 256, P3);
    bn_relu_k<<<NB((long)B * 256 * P3), 256, 0, stream>>>(fu1, mean, istd, fbn1_g, fbn1_b, B, 256, P3);

    conv2d_k<<<NB((long)B * 128 * P3), 256, 0, stream>>>(fu1, fw2, fb2, fu2, B, 256, H3, H3, 128, 3, 1);
    bn_stats_k<<<128, BNT, 0, stream>>>(fu2, mean, istd, B, 128, P3);
    bn_relu_k<<<NB((long)B * 128 * P3), 256, 0, stream>>>(fu2, mean, istd, fbn2_g, fbn2_b, B, 128, P3);

    conv2d_k<<<NB((long)B * 64 * P3), 256, 0, stream>>>(fu2, fw3, fb3, fu3, B, 128, H3, H3, 64, 3, 1);
    bn_stats_k<<<64, BNT, 0, stream>>>(fu3, mean, istd, B, 64, P3);
    bn_relu_k<<<NB((long)B * 64 * P3), 256, 0, stream>>>(fu3, mean, istd, fbn3_g, fbn3_b, B, 64, P3);

    blend_k<<<NB((long)B * P3), 256, 0, stream>>>(fu3, bl_w, bl_b, blm, B, 64, P3);
    resize_k<<<NB((long)B * H0 * H0), 256, 0, stream>>>(blm, w_o, B, H3, H3, H0, H0);
    final_k<<<NB((long)B * 3 * H0 * H0), 256, 0, stream>>>(img1, img2, w_o, res_o, B, 3, H0 * H0);
}
